// transformer_block_34411277975654
// MI455X (gfx1250) — compile-verified, matrix operand order checked by reading, not run on hardware
//
#include <hip/hip_runtime.h>
#include <hip/hip_bf16.h>

// Problem constants (from reference)
#define B_   2
#define S_   2048
#define D_   1024
#define H_   16
#define P_   128
#define DH_  64
#define NS_  (S_ + P_)     // 2176
#define ROWS_ (B_ * S_)    // 4096

typedef __attribute__((ext_vector_type(8)))  __bf16 v8bf;
typedef __attribute__((ext_vector_type(16))) __bf16 v16bf;
typedef __attribute__((ext_vector_type(8)))  float  v8f;
typedef int v4i_g __attribute__((vector_size(16)));   // gcc-style, matches builtin proto

// ---- gfx1250 async global->LDS support (ASYNCcnt path), probed at compile time
#if defined(__HIP_DEVICE_COMPILE__) && defined(__has_builtin)
#  if __has_builtin(__builtin_amdgcn_global_load_async_to_lds_b128)
#    define HAVE_ASYNC_LDS 1
#  else
#    define HAVE_ASYNC_LDS 0
#  endif
#  if __has_builtin(__builtin_amdgcn_s_wait_asynccnt)
#    define HAVE_WAIT_ASYNC 1
#  else
#    define HAVE_WAIT_ASYNC 0
#  endif
#else
#  define HAVE_ASYNC_LDS 0
#  define HAVE_WAIT_ASYNC 0
#endif

#define AS1 __attribute__((address_space(1)))
#define AS3 __attribute__((address_space(3)))

__device__ __forceinline__ void async_b128(const void* g, unsigned short* l) {
#if HAVE_ASYNC_LDS
  // proto: (v4i addrspace(1)* src, v4i addrspace(3)* dst, imm i32 offset, imm i32 cpol)
  __builtin_amdgcn_global_load_async_to_lds_b128((AS1 v4i_g*)g, (AS3 v4i_g*)l, 0, 0);
#else
  (void)g; (void)l;
#endif
}
__device__ __forceinline__ void wait_async0() {
#if HAVE_WAIT_ASYNC
  __builtin_amdgcn_s_wait_asynccnt(0);
#elif defined(__HIP_DEVICE_COMPILE__)
  asm volatile("s_wait_asynccnt 0x0" ::: "memory");
#endif
}

__device__ __forceinline__ unsigned short f2bf(float f) {
  unsigned int u = __float_as_uint(f);
  u += 0x7fffu + ((u >> 16) & 1u);          // round-to-nearest-even
  return (unsigned short)(u >> 16);
}
__device__ __forceinline__ float gelu_f(float x) {
  const float c = 0.7978845608028654f;      // sqrt(2/pi)
  float t = tanhf(c * (x + 0.044715f * x * x * x));
  return 0.5f * x * (1.0f + t);
}

// ---------------------------------------------------------------- LayerNorm
__global__ __launch_bounds__(256)
void ln_rows(const float* __restrict__ x, const float* __restrict__ g,
             const float* __restrict__ b, unsigned short* __restrict__ out) {
  __shared__ float red[256];
  const int row = blockIdx.x, t = threadIdx.x;
  const float* xr = x + (long long)row * D_;
  float s = 0.f;
  for (int i = t; i < D_; i += 256) s += xr[i];
  red[t] = s; __syncthreads();
  for (int w = 128; w > 0; w >>= 1) { if (t < w) red[t] += red[t + w]; __syncthreads(); }
  float mean = red[0] * (1.0f / D_); __syncthreads();
  float v = 0.f;
  for (int i = t; i < D_; i += 256) { float d = xr[i] - mean; v += d * d; }
  red[t] = v; __syncthreads();
  for (int w = 128; w > 0; w >>= 1) { if (t < w) red[t] += red[t + w]; __syncthreads(); }
  float rstd = rsqrtf(red[0] * (1.0f / D_) + 1e-3f);
  unsigned short* orow = out + (long long)row * D_;
  for (int i = t; i < D_; i += 256)
    orow[i] = f2bf((xr[i] - mean) * rstd * g[i] + b[i]);
}

// ---------------------- weight f32(KxN) -> bf16(NxK) transposed conversion
// 32x32 LDS tile; reads coalesced along N, writes coalesced along K.
__global__ __launch_bounds__(256)
void cvt_transpose(const float* __restrict__ w, unsigned short* __restrict__ wt,
                   int K, int N) {
  __shared__ float tile[32][33];
  const int n0 = blockIdx.x * 32, k0 = blockIdx.y * 32;
  const int tn = threadIdx.x & 31, tr = threadIdx.x >> 5;   // 8 rows per pass
  for (int r = tr; r < 32; r += 8)
    tile[r][tn] = w[(long long)(k0 + r) * N + n0 + tn];
  __syncthreads();
  for (int r = tr; r < 32; r += 8)
    wt[(long long)(n0 + r) * K + k0 + tn] = f2bf(tile[tn][r]);
}

// ------------------------------------------------------------- WMMA GEMM
// C = A(MxK) * B, with B stored row-major N x K (pre-transposed weights / K,V^T).
// A_F32: A is f32 in memory, converted to bf16 while staging to LDS (sync path).
// EPI  : 1 = f32 out + bias
//        2 = bf16 out + bias + GELU
//        3 = bf16 out (attention A*V)
//        4 = f32 out, *scale then causal mask (scores)
// Block 128x128x64, 256 threads = 8 waves, wave tile 32x64 (2x4 wmma frags,
// 2 k-frags per tile => 16 WMMA per barrier). Double-buffered LDS; bf16 tiles
// staged with GLOBAL_LOAD_ASYNC_TO_LDS_B128 (ASYNCcnt).
// NOTE: all instantiations have M % 128 == 0 and K % 64 == 0; B rows clamped
// when N < 128 (A*V), stores masked by gcol < N.
template <bool A_F32, int EPI>
__global__ __launch_bounds__(256)
void gemm_wmma(const void* __restrict__ Av, const unsigned short* __restrict__ Bw,
               const float* __restrict__ bias,
               float* __restrict__ Cf, unsigned short* __restrict__ Cb,
               int M, int N, int K, int lda, int ldb, int ldc,
               long long zsA, long long zsB,
               long long zsC_h, long long zsC_b, int Hdiv,
               float scale, int maskP) {
  __shared__ __align__(16) unsigned short As[2][128 * 72]; // 128 rows x (64 + 8 pad)
  __shared__ __align__(16) unsigned short Bs[2][128 * 72]; // [n][k]

  const int z  = blockIdx.z;
  const int t  = threadIdx.x;
  const int m0 = blockIdx.y * 128;
  const int n0 = blockIdx.x * 128;

  const unsigned short* Ab = A_F32 ? nullptr : (const unsigned short*)Av + (long long)z * zsA;
  const float*          Af = A_F32 ? (const float*)Av + (long long)z * zsA : nullptr;
  const unsigned short* Bb = Bw + (long long)z * zsB;
  const long long coff = (long long)(z / Hdiv) * zsC_b + (long long)(z % Hdiv) * zsC_h;

  v8f acc[2][4];
  for (int i = 0; i < 2; ++i)
    for (int j = 0; j < 4; ++j)
      for (int e = 0; e < 8; ++e) acc[i][j][e] = 0.0f;

  const int lane = t & 31;
  const int wave = t >> 5;
  const int wm = (wave & 3) * 32;  // wave row offset in block tile
  const int wn = (wave >> 2) * 64; // wave col offset in block tile

  const int srow = t >> 1;         // staging: 128 rows, 2 threads per row
  const int sq   = (t & 1) * 32;   // 32-element half of the 64-wide k slice

#if HAVE_ASYNC_LDS
  constexpr bool asyncStage = true;
#else
  constexpr bool asyncStage = false;
#endif
  constexpr bool anyAsync = asyncStage;

  auto stageA = [&](int k0, unsigned short* buf) {
    unsigned short* dst = buf + srow * 72 + sq;
    if (!A_F32) {
      const unsigned short* src = Ab + (long long)(m0 + srow) * lda + k0 + sq;
      if (asyncStage) {
        async_b128(src, dst);           async_b128(src + 8, dst + 8);
        async_b128(src + 16, dst + 16); async_b128(src + 24, dst + 24);
      } else {
        for (int c = 0; c < 4; ++c) *(uint4*)(dst + 8 * c) = *(const uint4*)(src + 8 * c);
      }
    } else {
      const float* src = Af + (long long)(m0 + srow) * lda + k0 + sq;
      const float4* s4 = (const float4*)src;
      for (int c = 0; c < 8; ++c) {
        float4 q = s4[c];
        dst[c * 4 + 0] = f2bf(q.x); dst[c * 4 + 1] = f2bf(q.y);
        dst[c * 4 + 2] = f2bf(q.z); dst[c * 4 + 3] = f2bf(q.w);
      }
      if (k0 + 64 < K) __builtin_prefetch(src + 64, 0, 1);
    }
  };
  auto stageB = [&](int k0, unsigned short* buf) {
    int gn = n0 + srow;
    gn = (gn < N) ? gn : (N - 1);        // clamp (no-op when N % 128 == 0)
    unsigned short* dst = buf + srow * 72 + sq;
    const unsigned short* src = Bb + (long long)gn * ldb + k0 + sq;
    if (asyncStage) {
      async_b128(src, dst);           async_b128(src + 8, dst + 8);
      async_b128(src + 16, dst + 16); async_b128(src + 24, dst + 24);
    } else {
      for (int c = 0; c < 4; ++c) *(uint4*)(dst + 8 * c) = *(const uint4*)(src + 8 * c);
    }
  };

  const int nk = K >> 6;
  stageA(0, As[0]);
  stageB(0, Bs[0]);

  for (int kt = 0; kt < nk; ++kt) {
    const int cur = kt & 1;
    if (anyAsync) wait_async0();     // this wave's DMA into buf[cur] done
    __syncthreads();                 // all waves' tiles visible
    if (kt + 1 < nk) {               // prefetch next tile into other buffer
      stageA((kt + 1) * 64, As[1 - cur]);
      stageB((kt + 1) * 64, Bs[1 - cur]);
    }

    // ---- build fragments (ISA 7.12.2 16-bit layout) and WMMA ----
    const int lh = lane >> 4;        // which K-half group this lane holds
    const int lm = lane & 15;        // row (A) / col (B) within fragment
    for (int f = 0; f < 2; ++f) {    // two 32-deep k-fragments per 64-wide tile
      const int kb = f * 32 + lh * 8;
      v16bf afr[2], bfr[4];
      for (int i = 0; i < 2; ++i) {
        const int base = (wm + i * 16 + lm) * 72 + kb;
        union { v16bf v; v8bf h[2]; } u;
        u.h[0] = *(const v8bf*)&As[cur][base];
        u.h[1] = *(const v8bf*)&As[cur][base + 16];
        afr[i] = u.v;
      }
      for (int j = 0; j < 4; ++j) {
        const int base = (wn + j * 16 + lm) * 72 + kb;
        union { v16bf v; v8bf h[2]; } u;
        u.h[0] = *(const v8bf*)&Bs[cur][base];
        u.h[1] = *(const v8bf*)&Bs[cur][base + 16];
        bfr[j] = u.v;
      }
      for (int i = 0; i < 2; ++i)
        for (int j = 0; j < 4; ++j)
          acc[i][j] = __builtin_amdgcn_wmma_f32_16x16x32_bf16(
              false, afr[i], false, bfr[j], (short)0, acc[i][j], false, false);
    }
  }

  // ---- epilogue / store ----
  const int lh   = lane >> 4;
  const int lcol = lane & 15;
  for (int i = 0; i < 2; ++i)
    for (int j = 0; j < 4; ++j)
      for (int r = 0; r < 8; ++r) {
        const int grow = m0 + wm + i * 16 + lh * 8 + r;
        const int gcol = n0 + wn + j * 16 + lcol;
        if (grow < M && gcol < N) {
          float vl = acc[i][j][r];
          if (EPI == 1 || EPI == 2) vl += bias[gcol];
          if (EPI == 2) vl = gelu_f(vl);
          if (EPI == 4) { vl *= scale; if (gcol > grow + maskP) vl = -1e10f; }
          const long long idx = coff + (long long)grow * ldc + gcol;
          if (EPI == 2 || EPI == 3) Cb[idx] = f2bf(vl);
          else                      Cf[idx] = vl;
        }
      }
}

// ------------------------------------------------------------- row softmax
__global__ __launch_bounds__(256)
void softmax_rows(float* __restrict__ w) {
  __shared__ float red[256];
  const long long row = blockIdx.x;            // B*H*S rows of length NS_
  float* wr = w + row * NS_;
  const int t = threadIdx.x;
  float mx = -3.4e38f;
  for (int i = t; i < NS_; i += 256) mx = fmaxf(mx, wr[i]);
  red[t] = mx; __syncthreads();
  for (int s = 128; s > 0; s >>= 1) { if (t < s) red[t] = fmaxf(red[t], red[t + s]); __syncthreads(); }
  mx = red[0]; __syncthreads();
  float sm = 0.f;
  for (int i = t; i < NS_; i += 256) { float e = __expf(wr[i] - mx); wr[i] = e; sm += e; }
  red[t] = sm; __syncthreads();
  for (int s = 128; s > 0; s >>= 1) { if (t < s) red[t] += red[t + s]; __syncthreads(); }
  const float inv = 1.0f / red[0];
  for (int i = t; i < NS_; i += 256) wr[i] *= inv;
}

// ---------------- split heads + K concat (NxK) + V^T concat (NxK) + present
__global__ __launch_bounds__(256)
void split_heads(const float* __restrict__ qkv, unsigned short* __restrict__ qb,
                 unsigned short* __restrict__ kc, unsigned short* __restrict__ vt,
                 float* __restrict__ present) {
  const long long i = (long long)blockIdx.x * 256 + threadIdx.x; // over B*S*D
  if (i >= (long long)ROWS_ * D_) return;
  const int d = (int)(i % D_);
  const long long bs = i / D_;
  const int s = (int)(bs % S_);
  const int b = (int)(bs / S_);
  const int h = d / DH_, dh = d % DH_;
  const float* r = qkv + bs * (3 * D_);
  const float q = r[d], k = r[D_ + d], v = r[2 * D_ + d];
  const long long hb = (long long)b * H_ + h;
  qb[(hb * S_ + s) * DH_ + dh]         = f2bf(q);
  kc[(hb * NS_ + P_ + s) * DH_ + dh]   = f2bf(k);   // [kpos][d]  (N x K for scores)
  vt[(hb * DH_ + dh) * NS_ + P_ + s]   = f2bf(v);   // [d][kpos]  (N x K for A*V)
  present[(((long long)(b * 2 + 0) * H_ + h) * S_ + s) * DH_ + dh] = k;
  present[(((long long)(b * 2 + 1) * H_ + h) * S_ + s) * DH_ + dh] = v;
}

__global__ __launch_bounds__(256)
void past_copy(const float* __restrict__ past, unsigned short* __restrict__ kc,
               unsigned short* __restrict__ vt) {
  const int i = blockIdx.x * 256 + threadIdx.x; // B*H*P*DH
  if (i >= B_ * H_ * P_ * DH_) return;
  const int dh = i % DH_;
  const int r  = i / DH_;
  const int j  = r % P_;
  const int bh = r / P_;
  const int h  = bh % H_;
  const int b  = bh / H_;
  const float k = past[(((long long)(b * 2 + 0) * H_ + h) * P_ + j) * DH_ + dh];
  const float v = past[(((long long)(b * 2 + 1) * H_ + h) * P_ + j) * DH_ + dh];
  kc[((long long)bh * NS_ + j) * DH_ + dh] = f2bf(k);
  vt[((long long)bh * DH_ + dh) * NS_ + j] = f2bf(v);
}

__global__ __launch_bounds__(256)
void add_f32(const float* __restrict__ a, const float* __restrict__ b,
             float* __restrict__ o, long long n) {
  const long long i = (long long)blockIdx.x * 256 + threadIdx.x;
  if (i < n) o[i] = a[i] + b[i];
}

// ---------------------------------------------------------------- launcher
extern "C" void kernel_launch(void* const* d_in, const int* in_sizes, int n_in,
                              void* d_out, int out_size, void* d_ws, size_t ws_size,
                              hipStream_t stream) {
  (void)in_sizes; (void)n_in; (void)out_size; (void)ws_size;
  const float* x       = (const float*)d_in[0];
  const float* past    = (const float*)d_in[1];
  const float* qkv_w   = (const float*)d_in[2];
  const float* qkv_b   = (const float*)d_in[3];
  const float* merge_w = (const float*)d_in[4];
  const float* merge_b = (const float*)d_in[5];
  const float* ln1_w   = (const float*)d_in[6];
  const float* ln1_b   = (const float*)d_in[7];
  const float* ln2_w   = (const float*)d_in[8];
  const float* ln2_b   = (const float*)d_in[9];
  const float* g1      = (const float*)d_in[10];
  const float* b1      = (const float*)d_in[11];
  const float* g2      = (const float*)d_in[12];
  const float* b2      = (const float*)d_in[13];

  float* out_xm      = (float*)d_out;
  float* out_present = out_xm + (long long)ROWS_ * D_;                  // B,2,H,S,DH
  float* out_w       = out_present + (long long)B_ * 2 * H_ * S_ * DH_; // B,H,S,NS

  char* ws = (char*)d_ws; size_t off = 0;
  auto alloc = [&](size_t bytes) { void* p = ws + off; off += (bytes + 255) & ~(size_t)255; return p; };

  unsigned short* h_bf   = (unsigned short*)alloc((size_t)ROWS_ * D_ * 2);
  unsigned short* wqkv   = (unsigned short*)alloc((size_t)D_ * 3 * D_ * 2);   // N x K
  unsigned short* wmrg   = (unsigned short*)alloc((size_t)D_ * D_ * 2);       // N x K
  unsigned short* wl1    = (unsigned short*)alloc((size_t)D_ * 4 * D_ * 2);   // N x K
  unsigned short* wl2    = (unsigned short*)alloc((size_t)4 * D_ * D_ * 2);   // N x K
  float*          qkv_f  = (float*)alloc((size_t)ROWS_ * 3 * D_ * 4);   // reused as t1/t2 later
  unsigned short* q_bf   = (unsigned short*)alloc((size_t)B_ * H_ * S_ * DH_ * 2);
  unsigned short* kc_bf  = (unsigned short*)alloc((size_t)B_ * H_ * NS_ * DH_ * 2);
  unsigned short* vt_bf  = (unsigned short*)alloc((size_t)B_ * H_ * NS_ * DH_ * 2);
  unsigned short* a_bf   = (unsigned short*)alloc((size_t)ROWS_ * D_ * 2);
  float*          xa_f   = (float*)alloc((size_t)ROWS_ * D_ * 4);
  unsigned short* h2_bf  = (unsigned short*)alloc((size_t)ROWS_ * D_ * 2);
  unsigned short* g_bf   = (unsigned short*)alloc((size_t)ROWS_ * 4 * D_ * 2);
  float* t1_f = qkv_f;                              // merge-proj out (qkv_f free by then)
  float* t2_f = qkv_f + (long long)ROWS_ * D_;      // mlp2 out

  const long long nXD = (long long)ROWS_ * D_;

  // 1. LN1 -> bf16
  ln_rows<<<ROWS_, 256, 0, stream>>>(x, g1, b1, h_bf);
  // 2. weight downconversion + transpose to N x K (one-time pass)
  cvt_transpose<<<dim3(96, 32),  256, 0, stream>>>(qkv_w,   wqkv, D_, 3 * D_);
  cvt_transpose<<<dim3(32, 32),  256, 0, stream>>>(merge_w, wmrg, D_, D_);
  cvt_transpose<<<dim3(128, 32), 256, 0, stream>>>(ln1_w,   wl1,  D_, 4 * D_);
  cvt_transpose<<<dim3(32, 128), 256, 0, stream>>>(ln2_w,   wl2,  4 * D_, D_);
  // 3. qkv = h @ qkv_w + b   (4096 x 3072 x 1024)
  gemm_wmma<false, 1><<<dim3(24, 32, 1), 256, 0, stream>>>(
      h_bf, wqkv, qkv_b, qkv_f, nullptr, ROWS_, 3 * D_, D_, D_, D_, 3 * D_,
      0, 0, 0, 0, 1, 1.0f, 0);
  // 4. split heads, K/V^T concat, present
  split_heads<<<(unsigned)(nXD / 256), 256, 0, stream>>>(qkv_f, q_bf, kc_bf, vt_bf, out_present);
  past_copy<<<(B_ * H_ * P_ * DH_) / 256, 256, 0, stream>>>(past, kc_bf, vt_bf);
  // 5. scores = scale*(Q @ K^T) with causal mask, straight into d_out  (per b,h)
  gemm_wmma<false, 4><<<dim3(17, 16, B_ * H_), 256, 0, stream>>>(
      q_bf, kc_bf, nullptr, out_w, nullptr, S_, NS_, DH_, DH_, DH_, NS_,
      (long long)S_ * DH_, (long long)NS_ * DH_,
      (long long)S_ * NS_, (long long)H_ * S_ * NS_, H_, 0.125f, P_);
  // 6. softmax in place on d_out
  softmax_rows<<<B_ * H_ * S_, 256, 0, stream>>>(out_w);
  // 7. A = w @ V  (f32 w -> bf16 in loader; B = V^T NxK), scatter into [b,s,h*64+dh]
  gemm_wmma<true, 3><<<dim3(1, 16, B_ * H_), 256, 0, stream>>>(
      out_w, vt_bf, nullptr, nullptr, a_bf, S_, DH_, NS_, NS_, NS_, D_,
      (long long)S_ * NS_, (long long)NS_ * DH_,
      (long long)DH_, (long long)S_ * D_, H_, 1.0f, 0);
  // 8. merge proj + residual
  gemm_wmma<false, 1><<<dim3(8, 32, 1), 256, 0, stream>>>(
      a_bf, wmrg, merge_b, t1_f, nullptr, ROWS_, D_, D_, D_, D_, D_,
      0, 0, 0, 0, 1, 1.0f, 0);
  add_f32<<<(unsigned)(nXD / 256), 256, 0, stream>>>(x, t1_f, xa_f, nXD);
  // 9. MLP: LN2 -> fc1+GELU (bf16 out) -> fc2 -> residual
  ln_rows<<<ROWS_, 256, 0, stream>>>(xa_f, g2, b2, h2_bf);
  gemm_wmma<false, 2><<<dim3(32, 32, 1), 256, 0, stream>>>(
      h2_bf, wl1, ln1_b, nullptr, g_bf, ROWS_, 4 * D_, D_, D_, D_, 4 * D_,
      0, 0, 0, 0, 1, 1.0f, 0);
  gemm_wmma<false, 1><<<dim3(8, 32, 1), 256, 0, stream>>>(
      g_bf, wl2, ln2_b, t2_f, nullptr, ROWS_, D_, 4 * D_, 4 * D_, 4 * D_, D_,
      0, 0, 0, 0, 1, 1.0f, 0);
  add_f32<<<(unsigned)(nXD / 256), 256, 0, stream>>>(x, t2_f, out_xm, nXD);
}
